// HGNNP_conv_implicit_63118839382184
// MI455X (gfx1250) — matrix operand order repaired and checked by reading, not compiled, not run on hardware
//
#include <hip/hip_runtime.h>

typedef __attribute__((ext_vector_type(16))) __bf16 v16bf;
typedef __attribute__((ext_vector_type(8)))  float  v8f;

union BF16x16 { v16bf v; unsigned u[8]; };

#define FT 128

// Split a pair of floats into (hi, lo) bf16x2 packed words.
// hi = truncate-to-bf16(x), lo = bf16(x - hi). Element 0 -> bits[15:0].
__device__ inline void split_pack(float a, float b, unsigned &hi, unsigned &lo) {
    unsigned ua = __float_as_uint(a), ub = __float_as_uint(b);
    unsigned ha = ua & 0xFFFF0000u, hb = ub & 0xFFFF0000u;
    float ra = a - __uint_as_float(ha);
    float rb = b - __uint_as_float(hb);
    hi = (ha >> 16) | hb;
    lo = ((__float_as_uint(ra) & 0xFFFF0000u) >> 16) |
         (__float_as_uint(rb) & 0xFFFF0000u);
}

// -------------------------------------------------------------------------
// GEMM: out[N x 128] = x[N x 128] @ W[128 x 128] + bias, split-bf16 WMMA.
// Block = 256 threads (8 waves). Each block: 16 rows; wave nt handles
// columns [16*nt, 16*nt+16). K = 128 = 4 k-tiles of 32.
// -------------------------------------------------------------------------
__global__ void __launch_bounds__(256) gemm_xw_bias(
    const float* __restrict__ x, const float* __restrict__ w,
    const float* __restrict__ bias, float* __restrict__ out, int N)
{
    extern __shared__ unsigned smem[];
    unsigned* sBhi = smem;                 // [4][8][8][32] = 8192
    unsigned* sBlo = smem + 8192;          // 8192
    unsigned* sAhi = smem + 16384;         // [16][65] = 1040 (padded rows)
    unsigned* sAlo = smem + 16384 + 1040;  // 1040

    const int t  = threadIdx.x;
    const int r0 = blockIdx.x * 16;

    // Stage W (hi/lo) as B operand, contiguous k order (not AMD's table order): VGPR j, lane l<16: K=32kt+2j, N=16nt+l
    //                                   lane l>=16:        K=32kt+16+2j, N=16nt+(l-16)
    for (int f = t; f < 8192; f += 256) {
        int lane = f & 31, v = (f >> 5) & 7, nt = (f >> 8) & 7, kt = f >> 11;
        int k = 32 * kt + ((lane >> 4) << 4) + 2 * v;
        int n = 16 * nt + (lane & 15);
        split_pack(w[k * FT + n], w[(k + 1) * FT + n], sBhi[f], sBlo[f]);
    }
    // Stage the 16xK tile of x as packed bf16 pairs: sA[m][pr] = k=2*pr,2*pr+1
    for (int p = t; p < 1024; p += 256) {
        int m = p >> 6, pr = p & 63, k = pr * 2;
        int row = r0 + m;
        float a0 = 0.f, a1 = 0.f;
        if (row < N) { a0 = x[row * FT + k]; a1 = x[row * FT + k + 1]; }
        split_pack(a0, a1, sAhi[m * 65 + pr], sAlo[m * 65 + pr]);
    }
    __syncthreads();

    const int lane = t & 31, nt = t >> 5;
    const int sub  = lane >> 4, m = lane & 15;
    v8f acc = {};

    for (int kt = 0; kt < 4; ++kt) {
        BF16x16 ah, al, bh, bl;
        #pragma unroll
        for (int v = 0; v < 8; ++v) {
            // A operand, contiguous k order (same as B): lane<16 holds K 0..15 (v0..7)
            //                           lane>=16 holds K 16..31
            int kk = 16 * sub + 2 * v;
            int pr = 16 * kt + (kk >> 1);
            ah.u[v] = sAhi[m * 65 + pr];
            al.u[v] = sAlo[m * 65 + pr];
            int bf = kt * 2048 + nt * 256 + v * 32 + lane;
            bh.u[v] = sBhi[bf];
            bl.u[v] = sBlo[bf];
        }
        acc = __builtin_amdgcn_wmma_f32_16x16x32_bf16(false, ah.v, false, bh.v,
                                                      (short)0, acc, false, false);
        acc = __builtin_amdgcn_wmma_f32_16x16x32_bf16(false, ah.v, false, bl.v,
                                                      (short)0, acc, false, false);
        acc = __builtin_amdgcn_wmma_f32_16x16x32_bf16(false, al.v, false, bh.v,
                                                      (short)0, acc, false, false);
    }

    // C/D layout: VGPR r, lane<16 -> M=r, N=lane; lane>=16 -> M=8+r, N=lane-16
    const int col  = nt * 16 + m;
    const float bc = bias[col];
    #pragma unroll
    for (int r = 0; r < 8; ++r) {
        int row = r0 + 8 * sub + r;
        if (row < N) out[row * FT + col] = acc[r] + bc;
    }
}

// -------------------------------------------------------------------------
// CSR construction helpers
// -------------------------------------------------------------------------
__global__ void zero_ints(int* __restrict__ p, int n) {
    int i = blockIdx.x * blockDim.x + threadIdx.x;
    if (i < n) p[i] = 0;
}

__global__ void count_deg(const int* __restrict__ ni, const int* __restrict__ ei,
                          int nnz, int* __restrict__ ndeg, int* __restrict__ edeg) {
    int i = blockIdx.x * blockDim.x + threadIdx.x;
    if (i < nnz) {
        atomicAdd(&edeg[ei[i]], 1);
        atomicAdd(&ndeg[ni[i]], 1);
    }
}

// Single-block exclusive scan (Hillis-Steele per 1024-chunk + carry).
__global__ void __launch_bounds__(1024) exscan(const int* __restrict__ in,
                                               int* __restrict__ out, int n) {
    __shared__ int sh[1024];
    __shared__ int carry_s;
    const int t = threadIdx.x;
    if (t == 0) carry_s = 0;
    __syncthreads();
    for (int base = 0; base < n; base += 1024) {
        int v = (base + t < n) ? in[base + t] : 0;
        sh[t] = v;
        __syncthreads();
        for (int off = 1; off < 1024; off <<= 1) {
            int xv = (t >= off) ? sh[t - off] : 0;
            __syncthreads();
            sh[t] += xv;
            __syncthreads();
        }
        int incl  = sh[t];
        int carry = carry_s;
        if (base + t < n) out[base + t] = carry + incl - v;
        __syncthreads();
        if (t == 1023) carry_s = carry + incl;
        __syncthreads();
    }
    if (t == 0) out[n] = carry_s;
}

__global__ void fill_csr(const int* __restrict__ ni, const int* __restrict__ ei,
                         const float* __restrict__ hv, int nnz,
                         const int* __restrict__ eoff, int* __restrict__ ecur,
                         int* __restrict__ enodes, float* __restrict__ evals,
                         const int* __restrict__ noff, int* __restrict__ ncur,
                         int* __restrict__ nedges, float* __restrict__ nvals) {
    int i = blockIdx.x * blockDim.x + threadIdx.x;
    if (i < nnz) {
        int e = ei[i], n = ni[i];
        float v = hv[i];
        int p = eoff[e] + atomicAdd(&ecur[e], 1);
        enodes[p] = n; evals[p] = v;
        int q = noff[n] + atomicAdd(&ncur[n], 1);
        nedges[q] = e; nvals[q] = v;
    }
}

// -------------------------------------------------------------------------
// Phase 2: one wave per hyperedge. E2[e] = de_inv[e] * sum_n xm[n]*dv_inv[n]*h
// Gathers from x_mapped (L2-resident). float4 per lane -> 512B per wave.
// -------------------------------------------------------------------------
__global__ void __launch_bounds__(256) edge_agg(
    const float* __restrict__ xm, const int* __restrict__ eoff,
    const int* __restrict__ enodes, const float* __restrict__ evals,
    const float* __restrict__ dv, const float* __restrict__ de,
    float* __restrict__ E2, int M)
{
    int wave = (int)((blockIdx.x * blockDim.x + threadIdx.x) >> 5);
    int lane = threadIdx.x & 31;
    if (wave >= M) return;
    int s = eoff[wave], e = eoff[wave + 1];
    float4 acc = make_float4(0.f, 0.f, 0.f, 0.f);
    const float4* xv = (const float4*)xm;
    for (int idx = s; idx < e; ++idx) {
        int nd   = enodes[idx];
        float sc = evals[idx] * dv[nd];
        float4 xr = xv[(size_t)nd * 32 + lane];
        acc.x += xr.x * sc; acc.y += xr.y * sc;
        acc.z += xr.z * sc; acc.w += xr.w * sc;
    }
    float d = de[wave];
    acc.x *= d; acc.y *= d; acc.z *= d; acc.w *= d;
    ((float4*)E2)[(size_t)wave * 32 + lane] = acc;
}

// -------------------------------------------------------------------------
// Phase 4 + epilogue: one wave per node.
// out[n] = x_mapped[n] (already in out) + dv_inv[n] * sum_e E2[e]*h
// -------------------------------------------------------------------------
__global__ void __launch_bounds__(256) node_agg_final(
    const float* __restrict__ E2, const int* __restrict__ noff,
    const int* __restrict__ nedges, const float* __restrict__ nvals,
    const float* __restrict__ dv, float* __restrict__ out, int N)
{
    int wave = (int)((blockIdx.x * blockDim.x + threadIdx.x) >> 5);
    int lane = threadIdx.x & 31;
    if (wave >= N) return;
    int s = noff[wave], e = noff[wave + 1];
    float4 acc = make_float4(0.f, 0.f, 0.f, 0.f);
    const float4* Ev = (const float4*)E2;
    for (int idx = s; idx < e; ++idx) {
        int ed   = nedges[idx];
        float sc = nvals[idx];
        float4 er = Ev[(size_t)ed * 32 + lane];
        acc.x += er.x * sc; acc.y += er.y * sc;
        acc.z += er.z * sc; acc.w += er.w * sc;
    }
    float d = dv[wave];
    float4* ov = (float4*)out;
    float4 cur = ov[(size_t)wave * 32 + lane];
    cur.x += acc.x * d; cur.y += acc.y * d;
    cur.z += acc.z * d; cur.w += acc.w * d;
    ov[(size_t)wave * 32 + lane] = cur;
}

// -------------------------------------------------------------------------
extern "C" void kernel_launch(void* const* d_in, const int* in_sizes, int n_in,
                              void* d_out, int out_size, void* d_ws, size_t ws_size,
                              hipStream_t stream) {
    const float* x        = (const float*)d_in[0];
    const float* w        = (const float*)d_in[1];
    const float* bias     = (const float*)d_in[2];
    const int*   node_idx = (const int*)d_in[3];
    const int*   edge_idx = (const int*)d_in[4];
    const float* h_val    = (const float*)d_in[5];
    const float* dv_inv   = (const float*)d_in[6];
    const float* de_inv   = (const float*)d_in[7];
    float* out = (float*)d_out;

    const int NNZ = in_sizes[3];
    const int N   = in_sizes[6];   // nodes
    const int M   = in_sizes[7];   // hyperedges

    // Workspace carve-out (256B aligned)
    char* p = (char*)d_ws;
    auto carve = [&](size_t bytes) -> char* {
        char* r = p;
        p += (bytes + 255) & ~(size_t)255;
        return r;
    };
    float* E2       = (float*)carve(sizeof(float) * (size_t)M * FT);
    int*   eoff     = (int*)  carve(sizeof(int) * (size_t)(M + 1));
    int*   noff     = (int*)  carve(sizeof(int) * (size_t)(N + 1));
    int*   zbase    = (int*)  carve(sizeof(int) * (size_t)(2 * M + 2 * N));
    int*   edeg = zbase;
    int*   ecur = edeg + M;
    int*   ndeg = ecur + M;
    int*   ncur = ndeg + N;
    int*   enodes   = (int*)  carve(sizeof(int)   * (size_t)NNZ);
    float* evals    = (float*)carve(sizeof(float) * (size_t)NNZ);
    int*   nedges   = (int*)  carve(sizeof(int)   * (size_t)NNZ);
    float* nvals    = (float*)carve(sizeof(float) * (size_t)NNZ);

    // 1) Build CSR both ways (int atomics only).
    int zn = 2 * M + 2 * N;
    zero_ints<<<(zn + 255) / 256, 256, 0, stream>>>(zbase, zn);
    count_deg<<<(NNZ + 255) / 256, 256, 0, stream>>>(node_idx, edge_idx, NNZ, ndeg, edeg);
    exscan<<<1, 1024, 0, stream>>>(edeg, eoff, M);
    exscan<<<1, 1024, 0, stream>>>(ndeg, noff, N);
    fill_csr<<<(NNZ + 255) / 256, 256, 0, stream>>>(node_idx, edge_idx, h_val, NNZ,
                                                    eoff, ecur, enodes, evals,
                                                    noff, ncur, nedges, nvals);

    // 2) GEMM with split-bf16 WMMA: out = x @ W + bias   (74 KB dynamic LDS)
    size_t shmem = (size_t)(8192 + 8192 + 1040 + 1040) * sizeof(unsigned);
    gemm_xw_bias<<<(N + 15) / 16, 256, shmem, stream>>>(x, w, bias, out, N);

    // 3) Edge-side segment sum (gather style, no float atomics).
    edge_agg<<<(M + 7) / 8, 256, 0, stream>>>(out, eoff, enodes, evals,
                                              dv_inv, de_inv, E2, M);

    // 4) Node-side segment sum + residual epilogue, in-place into out.
    node_agg_final<<<(N + 7) / 8, 256, 0, stream>>>(E2, noff, nedges, nvals,
                                                    dv_inv, out, N);
}
